// NonLocalBlockLinear_67482526155125
// MI455X (gfx1250) — compile-verified, matrix operand order checked by reading, not run on hardware
//
#include <hip/hip_runtime.h>

// ---------------------------------------------------------------------------
// NonLocal block (embedded-gaussian self-attention) for MI455X / gfx1250.
// B=8, C=512, H=W=56 (N=3136), IC=256.
// All GEMMs on v_wmma_f32_16x16x32_bf16; attention is flash-style (online
// softmax); K/V tiles staged with GLOBAL_LOAD_ASYNC_TO_LDS_B128 + ASYNCcnt.
// sched_group_barrier pins DS-read / WMMA interleaving (CK-style) so LDS
// latency overlaps the matrix pipe instead of serializing on s_wait_dscnt 0.
// ---------------------------------------------------------------------------

#define Bn   8
#define Cc   512
#define ICc  256
#define Nn   3136          // 56*56 = 49*64

typedef __attribute__((ext_vector_type(16))) __bf16 bf16x16;
typedef __attribute__((ext_vector_type(8)))  float  floatx8;

__device__ __forceinline__ floatx8 wmma_bf16(bf16x16 a, bf16x16 b, floatx8 c) {
  return __builtin_amdgcn_wmma_f32_16x16x32_bf16(
      false, a, false, b, (short)0, c, false, false);
}

// Build a 16x32 A/B fragment from a row pointer (wave32 WMMA K-pattern:
// halves 0..7 -> kb..kb+7, halves 8..15 -> 16+kb..16+kb+7; kb = 8*(lane>=16)).
template <typename T>
__device__ __forceinline__ bf16x16 ld_frag(const T* rowp, int kb) {
  bf16x16 f;
#pragma unroll
  for (int h = 0; h < 8; ++h) {
    f[h]     = rowp[kb + h];
    f[h + 8] = rowp[16 + kb + h];
  }
  return f;
}

// async 16-byte global -> LDS copy (tracked by ASYNCcnt)
__device__ __forceinline__ void async_copy_b128(unsigned lds_byte, const void* g) {
  asm volatile("global_load_async_to_lds_b128 %0, %1, off"
               :: "v"(lds_byte), "v"(g) : "memory");
}

// Pin schedule: FRONT ds_reads, then [1 WMMA, 2 ds_reads] x PAIRS, then 2 WMMA.
// mask 0x100 = DS read, 0x008 = MFMA/WMMA.  Best-effort, never unsafe.
#define SCHED_INTERLEAVE(FRONT, PAIRS)                              \
  do {                                                              \
    __builtin_amdgcn_sched_group_barrier(0x100, (FRONT), 0);        \
    _Pragma("unroll")                                               \
    for (int _i = 0; _i < (PAIRS); ++_i) {                          \
      __builtin_amdgcn_sched_group_barrier(0x008, 1, 0);            \
      __builtin_amdgcn_sched_group_barrier(0x100, 2, 0);            \
    }                                                               \
    __builtin_amdgcn_sched_group_barrier(0x008, 2, 0);              \
  } while (0)

// ===========================================================================
// Kernel 1: Q/K/V projections:  out[n,i] = sum_c x[b,c,n] * w[i,c] + bias[i]
// grid (49, 8, 3) ; block 256 (8 waves).  Block: 64 tokens x 256 IC.
// Wave w: rt=w&3 (16 tokens), ch=w>>2 (128 IC).  g (p==2) stores V^T [B,IC,N].
// ===========================================================================
__global__ __launch_bounds__(256) void qkv_kernel(
    const float* __restrict__ x,
    const float* __restrict__ theta_w, const float* __restrict__ theta_b,
    const float* __restrict__ phi_w,   const float* __restrict__ phi_b,
    const float* __restrict__ g_w,     const float* __restrict__ g_b,
    __bf16* __restrict__ Q, __bf16* __restrict__ K, __bf16* __restrict__ Vt)
{
  __shared__ __bf16 lx[64][32];    // [token][c]   4 KB
  __shared__ __bf16 lw[256][32];   // [ic][c]     16 KB

  const int tt = blockIdx.x, b = blockIdx.y, p = blockIdx.z;
  const float* w    = (p == 0) ? theta_w : (p == 1) ? phi_w : g_w;
  const float* bias = (p == 0) ? theta_b : (p == 1) ? phi_b : g_b;

  const int tid  = threadIdx.x;
  const int lane = tid & 31, wv = tid >> 5;
  const int lhi  = lane >> 4, lo = lane & 15, kb = lhi * 8;
  const int rt   = wv & 3, ch = wv >> 2;
  const int n0   = tt * 64;
  const int stok = tid & 63, scb = tid >> 6;

  floatx8 acc[8];
#pragma unroll
  for (int i = 0; i < 8; ++i)
#pragma unroll
    for (int e = 0; e < 8; ++e) acc[i][e] = 0.f;

  for (int c0 = 0; c0 < Cc; c0 += 32) {
    __syncthreads();
    // stage x tile 64x32 (register-buffered: all loads issued, one wait)
    {
      const float* xp = &x[((long)b * Cc + c0 + scb) * Nn + n0 + stok];
      float xr[8];
#pragma unroll
      for (int s = 0; s < 8; ++s) xr[s] = xp[(long)(4 * s) * Nn];
#pragma unroll
      for (int s = 0; s < 8; ++s) lx[stok][scb + 4 * s] = (__bf16)xr[s];
    }
    // stage weight tile 256x32 (one ic row per thread, 8x b128 loads)
    {
      const float* wp = &w[(long)tid * Cc + c0];
      float wr[32];
#pragma unroll
      for (int s = 0; s < 32; ++s) wr[s] = wp[s];
#pragma unroll
      for (int s = 0; s < 32; ++s) lw[tid][s] = (__bf16)wr[s];
    }
    __syncthreads();

    bf16x16 a = ld_frag(&lx[rt * 16 + lo][0], kb);
    bf16x16 f0 = ld_frag(&lw[ch * 128 + lo][0], kb);
    bf16x16 f1 = ld_frag(&lw[ch * 128 + 16 + lo][0], kb);
#pragma unroll
    for (int ct = 0; ct < 8; ++ct) {
      bf16x16 fn;
      if (ct + 2 < 8) fn = ld_frag(&lw[ch * 128 + (ct + 2) * 16 + lo][0], kb);
      acc[ct] = wmma_bf16(a, f0, acc[ct]);
      f0 = f1;
      f1 = fn;
    }
    SCHED_INTERLEAVE(6, 6);   // 18 ds reads, 8 WMMA
  }

#pragma unroll
  for (int ct = 0; ct < 8; ++ct) {
    int ic = ch * 128 + ct * 16 + lo;
    float bv = bias[ic];
#pragma unroll
    for (int r = 0; r < 8; ++r) {
      int tok = n0 + rt * 16 + r + 8 * lhi;
      float v = acc[ct][r] + bv;
      if (p == 0)      Q [((long)b * Nn + tok) * ICc + ic] = (__bf16)v;
      else if (p == 1) K [((long)b * Nn + tok) * ICc + ic] = (__bf16)v;
      else             Vt[((long)b * ICc + ic) * Nn + tok] = (__bf16)v;
    }
  }
}

// ===========================================================================
// Kernel 2: flash attention.  grid (49, 8) ; block 128 (4 waves).
// Wave = 16 query rows, all 3136 keys in chunks of 32.  Q frags in registers.
// K tile double-buffered + V tile staged via global_load_async_to_lds_b128.
// ASYNCcnt (in-order): oldest = lk[cur](8), then lv(8), then lk[next](8).
// LDS: 2x16K (K) + 16K (V^T) + 4K (P) = 52 KB.
// ===========================================================================
__global__ __launch_bounds__(128) void attn_kernel(
    const __bf16* __restrict__ Qg, const __bf16* __restrict__ Kg,
    const __bf16* __restrict__ Vt, __bf16* __restrict__ Y)
{
  __shared__ __bf16 lk[2][32][256];  // [buf][key][ic]  32 KB
  __shared__ __bf16 lv[256][32];     // [ic][key]       16 KB
  __shared__ __bf16 lp[4][16][32];   // per-wave P       4 KB

  const int qt = blockIdx.x, b = blockIdx.y;
  const int tid = threadIdx.x, lane = tid & 31, wv = tid >> 5;
  const int lhi = lane >> 4, lo = lane & 15, kb = lhi * 8;
  const int n0  = qt * 64;
  const int row = n0 + wv * 16 + lo;

  const unsigned lk_base = (unsigned)(uintptr_t)&lk[0][0][0];
  const unsigned lv_base = (unsigned)(uintptr_t)&lv[0][0];

  // preload Q fragments (8 K-chunks) into registers
  bf16x16 qf[8];
#pragma unroll
  for (int kc = 0; kc < 8; ++kc)
    qf[kc] = ld_frag(Qg + ((long)b * Nn + row) * ICc + kc * 32, kb);

  float rm[8], rl[8];
#pragma unroll
  for (int r = 0; r < 8; ++r) { rm[r] = -1e30f; rl[r] = 0.f; }
  floatx8 acc[16];
#pragma unroll
  for (int i = 0; i < 16; ++i)
#pragma unroll
    for (int e = 0; e < 8; ++e) acc[i][e] = 0.f;

  // issue async K tile for chunk 0 into buffer 0 (8 x b128 per wave)
  {
    const char* gK = (const char*)(Kg + ((long)b * Nn + 0) * ICc);
#pragma unroll
    for (int s = 0; s < 8; ++s) {
      unsigned u = tid + 128 * s;
      async_copy_b128(lk_base + u * 16u, gK + u * 16u);
    }
  }

  for (int j = 0, it = 0; j < Nn; j += 32, ++it) {
    const int  cur     = it & 1;
    const bool hasnext = (j + 32 < Nn);

    __syncthreads();   // everyone done with lv and lk[cur^1] from previous iter

    // async V^T tile for this chunk (overwrites lv -- safe after barrier)
#pragma unroll
    for (int s = 0; s < 8; ++s) {
      unsigned u = tid + 128 * s;
      unsigned ic = u >> 2, q = u & 3;
      const __bf16* g = Vt + ((long)b * ICc + ic) * Nn + j + q * 8;
      async_copy_b128(lv_base + ic * 64u + q * 16u, g);
    }
    // async K tile prefetch for next chunk into the other buffer
    if (hasnext) {
      const char* gK = (const char*)(Kg + ((long)b * Nn + j + 32) * ICc);
#pragma unroll
      for (int s = 0; s < 8; ++s) {
        unsigned u = tid + 128 * s;
        async_copy_b128(lk_base + (cur ^ 1) * 16384u + u * 16u, gK + u * 16u);
      }
      asm volatile("s_wait_asynccnt 16" ::: "memory");  // lk[cur] landed
    } else {
      asm volatile("s_wait_asynccnt 8" ::: "memory");   // lk[cur] landed
    }
    __syncthreads();   // lk[cur] visible to all waves

    // ---- S = Q K^T : 16 x 32, pipelined K fragments ----
    floatx8 s0, s1;
#pragma unroll
    for (int e = 0; e < 8; ++e) { s0[e] = 0.f; s1[e] = 0.f; }
    bf16x16 kf0 = ld_frag(&lk[cur][lo][0], kb);
    bf16x16 kf1 = ld_frag(&lk[cur][16 + lo][0], kb);
#pragma unroll
    for (int kc = 0; kc < 8; ++kc) {
      bf16x16 nf0, nf1;
      if (kc + 1 < 8) {
        nf0 = ld_frag(&lk[cur][lo][(kc + 1) * 32], kb);
        nf1 = ld_frag(&lk[cur][16 + lo][(kc + 1) * 32], kb);
      }
      s0 = wmma_bf16(qf[kc], kf0, s0);
      s1 = wmma_bf16(qf[kc], kf1, s1);
      kf0 = nf0;
      kf1 = nf1;
    }
    SCHED_INTERLEAVE(4, 14);  // 32 ds reads, 16 WMMA

    // ---- online softmax (rows r+8*lhi across the 16 lanes of each half) ----
    float nm[8];
#pragma unroll
    for (int r = 0; r < 8; ++r) nm[r] = fmaxf(s0[r], s1[r]);
#pragma unroll
    for (int m = 1; m < 16; m <<= 1)
#pragma unroll
      for (int r = 0; r < 8; ++r) nm[r] = fmaxf(nm[r], __shfl_xor(nm[r], m, 32));
#pragma unroll
    for (int r = 0; r < 8; ++r) nm[r] = fmaxf(nm[r], rm[r]);

    float sc[8], ps[8];
#pragma unroll
    for (int r = 0; r < 8; ++r) {
      sc[r] = __expf(rm[r] - nm[r]);
      s0[r] = __expf(s0[r] - nm[r]);
      s1[r] = __expf(s1[r] - nm[r]);
      ps[r] = s0[r] + s1[r];
    }
#pragma unroll
    for (int m = 1; m < 16; m <<= 1)
#pragma unroll
      for (int r = 0; r < 8; ++r) ps[r] += __shfl_xor(ps[r], m, 32);
#pragma unroll
    for (int r = 0; r < 8; ++r) { rl[r] = rl[r] * sc[r] + ps[r]; rm[r] = nm[r]; }
#pragma unroll
    for (int i = 0; i < 16; ++i)
#pragma unroll
      for (int r = 0; r < 8; ++r) acc[i][r] *= sc[r];

    // ---- P: D-layout -> A-layout via LDS bounce ----
#pragma unroll
    for (int r = 0; r < 8; ++r) {
      lp[wv][r + 8 * lhi][lo]      = (__bf16)s0[r];
      lp[wv][r + 8 * lhi][16 + lo] = (__bf16)s1[r];
    }

    if (hasnext) asm volatile("s_wait_asynccnt 8" ::: "memory");  // lv landed
    else         asm volatile("s_wait_asynccnt 0" ::: "memory");
    __syncthreads();   // lv + lp visible

    bf16x16 pa = ld_frag(&lp[wv][lo][0], kb);
    // ---- acc += P * V, pipelined V fragments ----
    bf16x16 vf = ld_frag(&lv[lo][0], kb);
#pragma unroll
    for (int ct = 0; ct < 16; ++ct) {
      bf16x16 nf;
      if (ct + 1 < 16) nf = ld_frag(&lv[(ct + 1) * 16 + lo][0], kb);
      acc[ct] = wmma_bf16(pa, vf, acc[ct]);
      vf = nf;
    }
    SCHED_INTERLEAVE(6, 14);  // 34 ds reads, 16 WMMA
  }

#pragma unroll
  for (int ct = 0; ct < 16; ++ct) {
#pragma unroll
    for (int r = 0; r < 8; ++r) {
      int tok = n0 + wv * 16 + r + 8 * lhi;
      int ic  = ct * 16 + lo;
      Y[((long)b * Nn + tok) * ICc + ic] = (__bf16)(acc[ct][r] / rl[r]);
    }
  }
}

// ===========================================================================
// Kernel 3: z-projection  y2[b,c,n] = sum_i Y[b,n,i]*wz_w[c,i] + wz_b[c]
// y2 stored [B,C,N] so this epilogue and the BN kernel are coalesced.
// grid (49, 8) ; block 256 (8 waves).
// ===========================================================================
__global__ __launch_bounds__(256) void zproj_kernel(
    const __bf16* __restrict__ Y, const float* __restrict__ wz_w,
    const float* __restrict__ wz_b, float* __restrict__ y2)
{
  __shared__ __bf16 ly[64][32];      // [token][ic]   4 KB
  __shared__ __bf16 lwz[512][32];    // [c][ic]      32 KB

  const int tt = blockIdx.x, b = blockIdx.y;
  const int tid = threadIdx.x, lane = tid & 31, wv = tid >> 5;
  const int lhi = lane >> 4, lo = lane & 15, kb = lhi * 8;
  const int rt = wv & 3, chalf = wv >> 2;
  const int n0 = tt * 64;

  floatx8 acc[16];
#pragma unroll
  for (int i = 0; i < 16; ++i)
#pragma unroll
    for (int e = 0; e < 8; ++e) acc[i][e] = 0.f;

  for (int kc = 0; kc < ICc; kc += 32) {
    __syncthreads();
    // stage Y tile 64x32 halfs as 256 x b128 copies (one per thread)
    {
      int tok = tid >> 2, q = tid & 3;
      const uint4* src =
          (const uint4*)&Y[((long)b * Nn + n0 + tok) * ICc + kc] + q;
      *(uint4*)&ly[tok][q * 8] = *src;
    }
    // stage wz tile 512x32 f32->bf16 (2 rows per thread)
#pragma unroll
    for (int rr = 0; rr < 2; ++rr) {
      int c = tid * 2 + rr;
      const float* wp = &wz_w[(long)c * ICc + kc];
      float wr[32];
#pragma unroll
      for (int s = 0; s < 32; ++s) wr[s] = wp[s];
#pragma unroll
      for (int s = 0; s < 32; ++s) lwz[c][s] = (__bf16)wr[s];
    }
    __syncthreads();

    bf16x16 a = ld_frag(&ly[rt * 16 + lo][0], kb);
    bf16x16 f0 = ld_frag(&lwz[chalf * 256 + lo][0], kb);
    bf16x16 f1 = ld_frag(&lwz[chalf * 256 + 16 + lo][0], kb);
#pragma unroll
    for (int ct = 0; ct < 16; ++ct) {
      bf16x16 fn;
      if (ct + 2 < 16)
        fn = ld_frag(&lwz[chalf * 256 + (ct + 2) * 16 + lo][0], kb);
      acc[ct] = wmma_bf16(a, f0, acc[ct]);
      f0 = f1;
      f1 = fn;
    }
    SCHED_INTERLEAVE(6, 14);  // 34 ds reads, 16 WMMA
  }

#pragma unroll
  for (int ct = 0; ct < 16; ++ct) {
    int c = chalf * 256 + ct * 16 + lo;
    float bv = wz_b[c];
#pragma unroll
    for (int r = 0; r < 8; ++r) {
      int tok = n0 + rt * 16 + r + 8 * lhi;          // 8 consecutive tokens
      y2[((long)b * Cc + c) * Nn + tok] = acc[ct][r] + bv;
    }
  }
}

// ===========================================================================
// Kernel 4: per-channel sum / sum-sq over (B,N) — deterministic reduction.
// ===========================================================================
__global__ __launch_bounds__(256) void stats_kernel(
    const float* __restrict__ y2, float* __restrict__ gsum,
    float* __restrict__ gsumsq)
{
  __shared__ float ss[256], sq[256];
  const int c = blockIdx.x, tid = threadIdx.x;
  float s = 0.f, q = 0.f;
  for (int b = 0; b < Bn; ++b) {
    const float* p = y2 + ((long)b * Cc + c) * Nn;
    for (int n = tid; n < Nn; n += 256) { float v = p[n]; s += v; q += v * v; }
  }
  ss[tid] = s; sq[tid] = q;
  __syncthreads();
  for (int o = 128; o > 0; o >>= 1) {
    if (tid < o) { ss[tid] += ss[tid + o]; sq[tid] += sq[tid + o]; }
    __syncthreads();
  }
  if (tid == 0) { gsum[c] = ss[0]; gsumsq[c] = sq[0]; }
}

// ===========================================================================
// Kernel 5: BatchNorm (batch stats, biased var) + affine + residual.
// Fully coalesced: y2, x, out all [B,C,N].
// ===========================================================================
__global__ __launch_bounds__(256) void bn_kernel(
    const float* __restrict__ y2, const float* __restrict__ x,
    const float* __restrict__ gsum, const float* __restrict__ gsumsq,
    const float* __restrict__ bn_w, const float* __restrict__ bn_b,
    float* __restrict__ out)
{
  long idx = (long)blockIdx.x * 256 + threadIdx.x;
  if (idx >= (long)Bn * Cc * Nn) return;
  int c = (int)((idx / Nn) % Cc);
  const float inv = 1.f / ((float)Bn * (float)Nn);
  float mean = gsum[c] * inv;
  float var  = gsumsq[c] * inv - mean * mean;
  float rs   = rsqrtf(var + 1e-5f);
  out[idx] = (y2[idx] - mean) * rs * bn_w[c] + bn_b[c] + x[idx];
}

// ===========================================================================
extern "C" void kernel_launch(void* const* d_in, const int* in_sizes, int n_in,
                              void* d_out, int out_size, void* d_ws, size_t ws_size,
                              hipStream_t stream) {
  const float* x       = (const float*)d_in[0];
  const float* theta_w = (const float*)d_in[1];
  const float* theta_b = (const float*)d_in[2];
  const float* phi_w   = (const float*)d_in[3];
  const float* phi_b   = (const float*)d_in[4];
  const float* g_w     = (const float*)d_in[5];
  const float* g_b     = (const float*)d_in[6];
  const float* wz_w    = (const float*)d_in[7];
  const float* wz_b    = (const float*)d_in[8];
  const float* bn_w    = (const float*)d_in[9];
  const float* bn_b    = (const float*)d_in[10];
  float* out = (float*)d_out;

  const size_t SZQ = (size_t)Bn * Nn * ICc;   // bf16 elements per Q/K/V/Y buf
  __bf16* Q  = (__bf16*)d_ws;
  __bf16* K  = Q + SZQ;
  __bf16* Vt = K + SZQ;
  __bf16* Y  = Vt + SZQ;
  float*  y2 = (float*)(Y + SZQ);
  float*  gsum   = y2 + (size_t)Bn * Nn * Cc;
  float*  gsumsq = gsum + Cc;

  qkv_kernel<<<dim3(49, 8, 3), 256, 0, stream>>>(
      x, theta_w, theta_b, phi_w, phi_b, g_w, g_b, Q, K, Vt);
  attn_kernel<<<dim3(49, 8), 128, 0, stream>>>(Q, K, Vt, Y);
  zproj_kernel<<<dim3(49, 8), 256, 0, stream>>>(Y, wz_w, wz_b, y2);
  stats_kernel<<<Cc, 256, 0, stream>>>(y2, gsum, gsumsq);
  bn_kernel<<<(int)(((long)Bn * Cc * Nn + 255) / 256), 256, 0, stream>>>(
      y2, x, gsum, gsumsq, bn_w, bn_b, out);
}
